// MADE_56014963474536
// MI455X (gfx1250) — matrix operand order repaired and checked by reading, not compiled, not run on hardware
//
#include <hip/hip_runtime.h>
#include <hip/hip_bf16.h>

// ---------------------------------------------------------------------------
// MADE autoregressive sampling, optimized for gfx1250 (MI455X):
//  - rank-1 incremental update of layer-0 preactivation (kills GEMM0)
//  - bf16 WMMA (v_wmma_f32_16x16x32_bf16) for the two 512x512 hidden GEMMs
//  - async global->LDS staging (ASYNCcnt) for weight/activation tiles
//  - masks applied analytically (deg_h = h % 31) -- no mask tensors needed
// ---------------------------------------------------------------------------

#define MD_D 32
#define MD_H 512
#define MD_B 2048

typedef __bf16 bf16;
typedef __attribute__((ext_vector_type(16))) __bf16 v16bf;
typedef __attribute__((ext_vector_type(8)))  __bf16 v8bf;
typedef __attribute__((ext_vector_type(8)))  float  v8f;

// --------------------- gfx1250 async global->LDS helpers -------------------
__device__ __forceinline__ void async_load_b128(const void* gaddr, unsigned lds_off) {
    // 16B per lane, global memory -> LDS, tracked with ASYNCcnt
    asm volatile("global_load_async_to_lds_b128 %0, %1, off"
                 :: "v"(lds_off), "v"(gaddr)
                 : "memory");
}
__device__ __forceinline__ void wait_async0() {
    asm volatile("s_wait_asynccnt 0x0" ::: "memory");
}

// ---------------------------------------------------------------------------
// Prep: masked bf16 copies of W1 and W2.  mask_h(r,c) = (r%31 >= c%31)
// ---------------------------------------------------------------------------
__global__ __launch_bounds__(256) void made_prep(const float* __restrict__ W1,
                                                 const float* __restrict__ W2,
                                                 bf16* __restrict__ MW1,
                                                 bf16* __restrict__ MW2) {
    int idx = blockIdx.x * 256 + threadIdx.x;          // 0 .. 512*512-1
    int r = idx >> 9;
    int c = idx & 511;
    float m = ((r % 31) >= (c % 31)) ? 1.0f : 0.0f;
    MW1[idx] = (bf16)(m * W1[idx]);
    MW2[idx] = (bf16)(m * W2[idx]);
}

// ---------------------------------------------------------------------------
// Init: h0_pre[b,h] = b0[h]  (x == 0 at step 0)
// ---------------------------------------------------------------------------
__global__ __launch_bounds__(256) void made_init(const float* __restrict__ b0,
                                                 float* __restrict__ h0_pre) {
    int idx = blockIdx.x * 256 + threadIdx.x;          // 0 .. B*H-1
    h0_pre[idx] = b0[idx & (MD_H - 1)];
}

// ---------------------------------------------------------------------------
// GEMM: Out[b,r] = relu( sum_c A[b,c] * W[r,c] + bias[r] ), Out in bf16.
// A is either f32 (relu + cvt fused into VGPR-path staging: layer-0 preact)
// or bf16 (staged async global->LDS).  W staged async global->LDS.
// Block: 128 threads = 4 waves; block tile 64(M) x 128(N); K staged 64-wide.
// Wave tile 32x64 = 2x4 tiles of v_wmma_f32_16x16x32_bf16.
// ---------------------------------------------------------------------------
#define TM 64
#define TN 128
#define TK 64
#define LP 8   // LDS row padding: (TK+LP)*2 = 144B pitch, keeps 16B alignment

template <bool A_F32_RELU>
__global__ __launch_bounds__(128)
void made_gemm(const void* __restrict__ Asrc,
               const bf16* __restrict__ W,
               const float* __restrict__ bias,
               bf16* __restrict__ Out) {
    __shared__ __align__(16) bf16 As[TM][TK + LP];
    __shared__ __align__(16) bf16 Bs[TN][TK + LP];

    const int tid   = threadIdx.x;
    const int lane  = tid & 31;
    const int wave  = tid >> 5;
    const int mBase = blockIdx.x * TM;
    const int nBase = blockIdx.y * TN;
    const int wm    = (wave >> 1) * 32;   // wave M offset (0 or 32)
    const int wn    = (wave & 1) * 64;    // wave N offset (0 or 64)

    v8f acc[2][4];
#pragma unroll
    for (int sm = 0; sm < 2; ++sm)
#pragma unroll
        for (int sn = 0; sn < 4; ++sn) acc[sm][sn] = (v8f)0.0f;

    for (int kt = 0; kt < MD_H; kt += TK) {
        // ---- stage B tile (128 n-rows x 32 k), async global->LDS ----
        {
            const bf16* src = W + (size_t)(nBase + tid) * MD_H + kt;
            const unsigned ldsb = (unsigned)(size_t)(&Bs[tid][0]);
#pragma unroll
            for (int e = 0; e < 4; ++e)
                async_load_b128(src + e * 8, ldsb + e * 16);
        }
        // ---- stage A tile (64 x 64): 2 threads per row, 32 elems each ----
        {
            const int r  = tid & 63;
            const int ch = tid >> 6;            // 0 or 1
            const int gr = mBase + r;
            if (A_F32_RELU) {
                // f32 source: fuse relu + f32->bf16 conversion (VGPR path)
                const float* src = (const float*)Asrc + (size_t)gr * MD_H + kt + ch * 32;
                bf16* dst = &As[r][ch * 32];
#pragma unroll
                for (int e = 0; e < 32; ++e) {
                    float v = src[e];
                    dst[e] = (bf16)(v > 0.0f ? v : 0.0f);
                }
            } else {
                // bf16 source: async global->LDS, no VGPR round trip
                const bf16* src = (const bf16*)Asrc + (size_t)gr * MD_H + kt + ch * 32;
                const unsigned ldsa = (unsigned)(size_t)(&As[r][ch * 32]);
#pragma unroll
                for (int e = 0; e < 4; ++e)
                    async_load_b128(src + e * 8, ldsa + e * 16);
            }
        }
        wait_async0();          // drain ASYNCcnt before making LDS visible
        __syncthreads();

        const int mrow = lane & 15;
        const int hi   = lane >> 4;
#pragma unroll
        for (int ks = 0; ks < TK; ks += 32) {
            union { v16bf v; v8bf h[2]; } af[2], bf[4];
            // A fragment: lane holds row M=lane%16; K chunks hi*8 and 16+hi*8
#pragma unroll
            for (int sm = 0; sm < 2; ++sm) {
                const bf16* p = &As[wm + sm * 16 + mrow][ks + hi * 8];
                af[sm].h[0] = *(const v8bf*)(p);
                af[sm].h[1] = *(const v8bf*)(p + 16);
            }
            // B fragment: lane holds col N=lane%16; K chunks hi*8 and 16+hi*8
#pragma unroll
            for (int sn = 0; sn < 4; ++sn) {
                const bf16* p = &Bs[wn + sn * 16 + mrow][ks + hi * 8];
                bf[sn].h[0] = *(const v8bf*)(p);
                bf[sn].h[1] = *(const v8bf*)(p + 16);
            }
#pragma unroll
            for (int sm = 0; sm < 2; ++sm)
#pragma unroll
                for (int sn = 0; sn < 4; ++sn)
                    acc[sm][sn] = __builtin_amdgcn_wmma_f32_16x16x32_bf16(
                        false, af[sm].v, false, bf[sn].v,
                        (short)0, acc[sm][sn], false, false);
        }
        __syncthreads();
    }

    // ---- epilogue: bias + relu + bf16 store ----
    const int col16 = lane & 15;
    const int rhi   = (lane >> 4) * 8;
#pragma unroll
    for (int sn = 0; sn < 4; ++sn) {
        const int gc = nBase + wn + sn * 16 + col16;
        const float bv = bias[gc];
#pragma unroll
        for (int sm = 0; sm < 2; ++sm) {
            const int gr0 = mBase + wm + sm * 16 + rhi;
#pragma unroll
            for (int v = 0; v < 8; ++v) {
                float x = acc[sm][sn][v] + bv;
                x = x > 0.0f ? x : 0.0f;
                Out[(size_t)(gr0 + v) * MD_H + gc] = (bf16)x;
            }
        }
    }
}

// ---------------------------------------------------------------------------
// Update: one wave per batch row.
//   mean_i   = sum_c h3[b,c]*mask_o(i,c)*Wout[i,c]      + bout[i]
//   logstd_i = sum_c h3[b,c]*mask_o(i,c)*Wout[32+i,c]   + bout[32+i]
//   x_i = mean_i + z[b,i]*exp(logstd_i)     -> written straight to d_out
//   h0_pre[b,:] += mask0(:,i) * W0[:,i] * x_i   (rank-1 update)
//   log_det accumulated in ws; flushed to d_out at i==31.
// mask_o(i,c) = (c%31 < i);  mask0(h,i) = (h%31 >= i)
// ---------------------------------------------------------------------------
__global__ __launch_bounds__(256)
void made_update(const bf16* __restrict__ h3, const float* __restrict__ Wout,
                 const float* __restrict__ bout, const float* __restrict__ z,
                 const float* __restrict__ W0, float* __restrict__ h0_pre,
                 float* __restrict__ ldws, float* __restrict__ out, int i) {
    const int wave = threadIdx.x >> 5;
    const int lane = threadIdx.x & 31;
    const int b = blockIdx.x * 8 + wave;

    const bf16* hrow = h3 + (size_t)b * MD_H;
    float s0 = 0.0f, s1 = 0.0f;
#pragma unroll
    for (int t = 0; t < 16; ++t) {
        const int c = lane + 32 * t;
        const float m  = ((c % 31) < i) ? 1.0f : 0.0f;
        const float hv = (float)hrow[c] * m;
        s0 += hv * Wout[(size_t)i * MD_H + c];
        s1 += hv * Wout[(size_t)(MD_D + i) * MD_H + c];
    }
#pragma unroll
    for (int off = 16; off > 0; off >>= 1) {
        s0 += __shfl_xor(s0, off, 32);
        s1 += __shfl_xor(s1, off, 32);
    }
    const float mean = s0 + bout[i];
    const float lstd = s1 + bout[MD_D + i];
    const float xi = mean + z[(size_t)b * MD_D + i] * __expf(lstd);
    const float ld = (i == 0 ? 0.0f : ldws[b]) - lstd;

    if (lane == 0) {
        out[(size_t)b * MD_D + i] = xi;       // x column i
        ldws[b] = ld;
        if (i == MD_D - 1) out[(size_t)MD_B * MD_D + b] = ld;  // log_det
    }
    if (i < MD_D - 1) {
#pragma unroll
        for (int t = 0; t < 16; ++t) {
            const int h = lane + 32 * t;
            const float m0 = ((h % 31) >= i) ? 1.0f : 0.0f;
            h0_pre[(size_t)b * MD_H + h] += m0 * W0[(size_t)h * MD_D + i] * xi;
        }
    }
}

// ---------------------------------------------------------------------------
extern "C" void kernel_launch(void* const* d_in, const int* in_sizes, int n_in,
                              void* d_out, int out_size, void* d_ws, size_t ws_size,
                              hipStream_t stream) {
    const float* z    = (const float*)d_in[0];
    const float* W0   = (const float*)d_in[1];
    const float* b0   = (const float*)d_in[2];
    const float* W1   = (const float*)d_in[3];
    const float* b1   = (const float*)d_in[4];
    const float* W2   = (const float*)d_in[5];
    const float* b2   = (const float*)d_in[6];
    const float* Wout = (const float*)d_in[7];
    const float* bout = (const float*)d_in[8];
    float* out = (float*)d_out;

    char* ws = (char*)d_ws;
    float* h0_pre = (float*)(ws);                              // 2048*512*4 = 4 MB
    bf16*  h2     = (bf16*) (ws + (size_t)4  * 1024 * 1024);   // 2 MB
    bf16*  h3     = (bf16*) (ws + (size_t)6  * 1024 * 1024);   // 2 MB
    bf16*  MW1    = (bf16*) (ws + (size_t)8  * 1024 * 1024);   // 512 KB
    bf16*  MW2    = (bf16*) (ws + (size_t)8  * 1024 * 1024 + 512 * 1024);
    float* ldws   = (float*)(ws + (size_t)9  * 1024 * 1024);   // 8 KB

    made_prep<<<(MD_H * MD_H) / 256, 256, 0, stream>>>(W1, W2, MW1, MW2);
    made_init<<<(MD_B * MD_H) / 256, 256, 0, stream>>>(b0, h0_pre);

    const dim3 ggrid(MD_B / TM, MD_H / TN);   // 32 x 4
    for (int i = 0; i < MD_D; ++i) {
        made_gemm<true ><<<ggrid, 128, 0, stream>>>((const void*)h0_pre, MW1, b1, h2);
        made_gemm<false><<<ggrid, 128, 0, stream>>>((const void*)h2,     MW2, b2, h3);
        made_update<<<MD_B / 8, 256, 0, stream>>>(h3, Wout, bout, z, W0,
                                                  h0_pre, ldws, out, i);
    }
}
